// DVGEDR_85693187489970
// MI455X (gfx1250) — compile-verified, matrix operand order checked by reading, not run on hardware
//
#include <hip/hip_runtime.h>
#include <hip/hip_bf16.h>

// ---------------------------------------------------------------------------
// Problem constants (match reference)
// ---------------------------------------------------------------------------
#define BGR   2048          // graphs
#define NPG   64            // nodes per graph
#define INF_  18            // input features
#define NT_   (BGR * NPG)   // 131072 nodes
#define EPG_  512           // edges per graph
#define NEDG  (EPG_ + NPG)  // edges + self loops per graph
#define KPOOL 30            // sort-pool k
#define LAT_  192           // concat latent
#define EPS_  1e-5f
#define SLOPE_ 0.2f

typedef __attribute__((ext_vector_type(16))) _Float16 v16h;
typedef __attribute__((ext_vector_type(8)))  _Float16 v8h;
typedef __attribute__((ext_vector_type(8)))  float    v8f;

// ---------- optional Tensor Data Mover path (toolchain-dependent) ----------
#if defined(__has_builtin)
#if __has_builtin(__builtin_amdgcn_tensor_load_to_lds) && \
    __has_builtin(__builtin_amdgcn_s_wait_tensorcnt)
#define HAVE_TDM 1
#endif
#endif
#ifndef HAVE_TDM
#define HAVE_TDM 0
#endif
#if __has_include(<hip/amd_detail/amd_gfx1250_TDM.h>)
#define TDM_SIX_ARG 1
#else
#define TDM_SIX_ARG 0
#endif

typedef __attribute__((ext_vector_type(4))) unsigned int u32x4;
typedef __attribute__((ext_vector_type(8))) int          i32x8;
typedef __attribute__((ext_vector_type(4))) int          i32x4;

// ---------------------------------------------------------------------------
// WMMA helpers (CDNA5 v_wmma_f32_16x16x32_f16)
// ---------------------------------------------------------------------------
__device__ __forceinline__ v8f wmma_f16(v16h a, v16h b, v8f c) {
  return __builtin_amdgcn_wmma_f32_16x16x32_f16(false, a, false, b,
                                                (short)0, c, false, false);
}

__device__ __forceinline__ v16h mk16(v8h lo, v8h hi) {
  v16h a;
#pragma unroll
  for (int e = 0; e < 8; ++e) { a[e] = lo[e]; a[e + 8] = hi[e]; }
  return a;
}

// A fragment (16x32 f16): lane l holds row M=l%16; element e -> k =
// (e>=8?16:0) + (l>=16?8:0) + (e&7).  Row-major f16 source: two contiguous
// 16-byte runs per lane -> two b128 loads.
__device__ __forceinline__ v16h ld_a_pk(const _Float16* Am, int row0, int k0,
                                        int ld, int lane) {
  const _Float16* p = Am + (row0 + (lane & 15)) * ld + k0 + ((lane >> 4) << 3);
  return mk16(*(const v8h*)p, *(const v8h*)(p + 16));
}

// B fragment (32x16 f16), from pre-packed column-major weights Wt[col*ldk+k]:
// same two contiguous 16-byte runs per lane.
__device__ __forceinline__ v16h ld_b_pk(const _Float16* Wt, int k0, int col0,
                                        int ldk, int lane) {
  const _Float16* p = Wt + (col0 + (lane & 15)) * ldk + k0 + ((lane >> 4) << 3);
  return mk16(*(const v8h*)p, *(const v8h*)(p + 16));
}

// A fragment from f32 LDS (compiler emits ds_load_b128 + v_cvt_pk_f16_f32)
__device__ __forceinline__ v16h ld_a_lds_f32(const float* s, int row0, int k0,
                                             int ld, int lane) {
  int r  = row0 + (lane & 15);
  int kb = k0 + ((lane >> 4) << 3);
  v16h a;
#pragma unroll
  for (int e = 0; e < 16; ++e) {
    int k = kb + ((e >= 8) ? 16 : 0) + (e & 7);
    a[e] = (_Float16)s[r * ld + k];
  }
  return a;
}

// C/D (16x16 f32): VGPR v, lane l -> row = v + (l>=16?8:0), col = l%16
__device__ __forceinline__ void st_c_lds(float* s, v8f c, int row0, int col0,
                                         int ld, int lane) {
  int cc = col0 + (lane & 15);
  int rb = row0 + ((lane >> 4) << 3);
#pragma unroll
  for (int v = 0; v < 8; ++v) s[(rb + v) * ld + cc] = c[v];
}

__device__ __forceinline__ v8f ld_c_lds(const float* s, int row0, int col0,
                                        int ld, int lane) {
  int cc = col0 + (lane & 15);
  int rb = row0 + ((lane >> 4) << 3);
  v8f c;
#pragma unroll
  for (int v = 0; v < 8; ++v) c[v] = s[(rb + v) * ld + cc];
  return c;
}

// monotone int encoding of floats so ds_max_i32 implements float max
__device__ __forceinline__ int fenc(float f) {
  int i = __float_as_int(f);
  return i < 0 ? (i ^ 0x7fffffff) : i;
}
__device__ __forceinline__ float fdec(int i) {
  return __int_as_float(i < 0 ? (i ^ 0x7fffffff) : i);
}

// ---------------------------------------------------------------------------
// Kp: pack a row-major f32 weight [K,N] into column-major f16 Wt[c*K + k],
// zero-padding rows k >= kvalid (removes all guards from B-fragment loads)
// ---------------------------------------------------------------------------
__global__ __launch_bounds__(256) void k_packw(const float* __restrict__ W,
                                               _Float16* __restrict__ Wt,
                                               int K, int N, int kvalid) {
  int i = blockIdx.x * 256 + threadIdx.x;
  if (i >= K * N) return;
  int c = i / K, k = i - c * K;
  Wt[i] = (_Float16)(k < kvalid ? W[k * N + c] : 0.f);
}

// ---------------------------------------------------------------------------
// K0: per-graph LayerNorm of raw input -> zero-padded [NT,64] tile buffer
// ---------------------------------------------------------------------------
__global__ __launch_bounds__(128) void k_initln(const float* __restrict__ x,
                                                const float* __restrict__ w,
                                                const float* __restrict__ b,
                                                float* __restrict__ xn) {
  __shared__ float red[128];
  __shared__ float smean, srs;
  const int g = blockIdx.x, tid = threadIdx.x;
  const int tot = NPG * INF_;
  float local = 0.f;
  for (int i = tid; i < tot; i += 128) local += x[g * tot + i];
  red[tid] = local; __syncthreads();
  for (int s = 64; s > 0; s >>= 1) { if (tid < s) red[tid] += red[tid + s]; __syncthreads(); }
  if (tid == 0) smean = red[0] / (float)tot;
  __syncthreads();
  const float m = smean;
  local = 0.f;
  for (int i = tid; i < tot; i += 128) { float d = x[g * tot + i] - m; local += d * d; }
  red[tid] = local; __syncthreads();
  for (int s = 64; s > 0; s >>= 1) { if (tid < s) red[tid] += red[tid + s]; __syncthreads(); }
  if (tid == 0) srs = rsqrtf(red[0] / (float)tot + EPS_);
  __syncthreads();
  const float r = srs;
  for (int i = tid; i < NPG * 64; i += 128) {
    int n = i >> 6, c = i & 63;
    float v = 0.f;
    if (c < INF_) v = (x[g * tot + n * INF_ + c] - m) * r * w[c] + b[c];
    xn[(g * NPG + n) * 64 + c] = v;
  }
}

// ---------------------------------------------------------------------------
// K1: fused GAT + LN + FFN + residual + LN block, one workgroup per graph
//   R0: x tile -> edge buckets -> FFN half activations
//   R1: h      -> FFN output accumulator u -> h2
//   R2: attention output h1 (residual)
// ---------------------------------------------------------------------------
struct GatArgs {
  float* x;            // [NT,64] in/out (in place per graph tile)
  float* cs;           // [NT,192] block outputs concatenated
  const int* esrc; const int* edst;
  const _Float16* gatWh;                       // packed [64c x 64k]
  const float* attS; const float* attD; const float* gatB;
  const float* ln1w; const float* ln1b;
  const _Float16* f1Wh;                        // packed [128c x 64k]
  const float* f1B;
  const _Float16* f2Wh;                        // packed [64c x 128k]
  const float* f2B;
  const float* ln2w; const float* ln2b;
  int blk;
};

__global__ __launch_bounds__(256) void k_gat(GatArgs A) {
  __shared__ __align__(16) float    R0[64 * 64];
  __shared__ __align__(16) float    R1[64 * 64];
  __shared__ __align__(16) float    R2[64 * 64];
  __shared__ __align__(16) _Float16 Ah[64 * 64];   // f16 LN1 output (WMMA A)
  __shared__ float sa[128], sd[128];
  __shared__ int   smax[128];
  __shared__ int   ecnt[64], eoff[64];
  __shared__ float mu[64], rs[64];

  const int g = blockIdx.x, tid = threadIdx.x;
  const int wave = tid >> 5, lane = tid & 31;
  const int base = g * NPG;

  // 1. load x tile into R0 (TDM if available, else vectorized copy)
#if HAVE_TDM
  if (tid == 0) {
    unsigned long long ga = (unsigned long long)(uintptr_t)(A.x + base * 64);
    unsigned int lds = (unsigned int)(unsigned long long)(uintptr_t)(void*)R0;
    u32x4 g0;
    g0[0] = 1u;                                       // count=1 descriptor
    g0[1] = lds;                                      // lds_addr
    g0[2] = (unsigned int)(ga & 0xffffffffu);         // global_addr lo
    g0[3] = (unsigned int)((ga >> 32) & 0x1ffffffu) | (2u << 30); // hi | type=2
    i32x8 g1;
    g1[0] = 0x20000;            // data_size=4B, wg_mask=0
    g1[1] = (int)(4096u << 16); // tensor_dim0 = 4096 elements
    g1[2] = 1 << 16;            // tensor_dim1 = 1
    g1[3] = (int)(4096u << 16); // tile_dim0 = 4096
    g1[4] = 1;                  // tile_dim1 = 1
    g1[5] = 4096;               // tensor_dim0_stride
    g1[6] = 0; g1[7] = 0;
    i32x4 g2; g2[0] = 0; g2[1] = 0; g2[2] = 0; g2[3] = 0;
    i32x4 g3 = g2;
#if TDM_SIX_ARG
    i32x8 g4; for (int q = 0; q < 8; ++q) g4[q] = 0;
    __builtin_amdgcn_tensor_load_to_lds(g0, g1, g2, g3, g4, 0);
#else
    __builtin_amdgcn_tensor_load_to_lds(g0, g1, g2, g3, 0);
#endif
  }
  __builtin_amdgcn_s_wait_tensorcnt((short)0);
#else
  for (int i = tid; i < 1024; i += 256)
    ((float4*)R0)[i] = ((const float4*)(A.x + base * 64))[i];
#endif
  __syncthreads();

  // 2. h = x @ gat_W  (WMMA, packed zero-padded f16 weights)
  for (int t = wave; t < 16; t += 8) {
    int tr = (t >> 2) << 4, tc = (t & 3) << 4;
    v8f c = {};
    for (int kk = 0; kk < 64; kk += 32) {
      v16h af = ld_a_lds_f32(R0, tr, kk, 64, lane);
      v16h bf = ld_b_pk(A.gatWh, kk, tc, 64, lane);
      c = wmma_f16(af, bf, c);
    }
    st_c_lds(R1, c, tr, tc, 64, lane);
  }
  __syncthreads();

  // 3. per-(node,head) attention scalars; init max and degree counters
  if (tid < 128) {
    int n = tid >> 1, hd = tid & 1;
    float s0 = 0.f, s1 = 0.f;
    for (int c = 0; c < 32; ++c) {
      float hv = R1[n * 64 + hd * 32 + c];
      s0 += hv * A.attS[hd * 32 + c];
      s1 += hv * A.attD[hd * 32 + c];
    }
    sa[tid] = s0; sd[tid] = s1;
    smax[tid] = (int)0x80000000;
  }
  if (tid < 64) ecnt[tid] = 0;
  __syncthreads();

  // 4. edge pass 1: segment max (ds_max_i32) + in-degree count
  for (int e = tid; e < NEDG; e += 256) {
    int sl, dl;
    if (e < EPG_) { sl = A.esrc[g * EPG_ + e] - base; dl = A.edst[g * EPG_ + e] - base; }
    else          { sl = dl = e - EPG_; }
    float v0 = sa[sl * 2] + sd[dl * 2];
    float v1 = sa[sl * 2 + 1] + sd[dl * 2 + 1];
    v0 = v0 > 0.f ? v0 : SLOPE_ * v0;
    v1 = v1 > 0.f ? v1 : SLOPE_ * v1;
    atomicMax(&smax[dl * 2], fenc(v0));
    atomicMax(&smax[dl * 2 + 1], fenc(v1));
    atomicAdd(&ecnt[dl], 1);
  }
  __syncthreads();
  if (tid == 0) {
    int run = 0;
    for (int n = 0; n < NPG; ++n) { eoff[n] = run; run += ecnt[n]; }
  }
  __syncthreads();
  if (tid < 64) ecnt[tid] = 0;
  __syncthreads();

  // 5. edge pass 2: bucket edges by destination (buffers alias dead R0)
  int*   eL = (int*)R0;          // [NEDG] source node per edge
  float* eE = R0 + 1024;         // [2*NEDG] exp weights per head
  for (int e = tid; e < NEDG; e += 256) {
    int sl, dl;
    if (e < EPG_) { sl = A.esrc[g * EPG_ + e] - base; dl = A.edst[g * EPG_ + e] - base; }
    else          { sl = dl = e - EPG_; }
    float v0 = sa[sl * 2] + sd[dl * 2];
    float v1 = sa[sl * 2 + 1] + sd[dl * 2 + 1];
    v0 = v0 > 0.f ? v0 : SLOPE_ * v0;
    v1 = v1 > 0.f ? v1 : SLOPE_ * v1;
    int pos = eoff[dl] + atomicAdd(&ecnt[dl], 1);
    eL[pos] = sl;
    eE[2 * pos]     = __expf(v0 - fdec(smax[dl * 2]));
    eE[2 * pos + 1] = __expf(v1 - fdec(smax[dl * 2 + 1]));
  }
  __syncthreads();

  // 6. gather-accumulate: thread = (node, 16-channel slice); no atomics
  {
    int n = tid >> 2, q = tid & 3;        // q: channels q*16 .. q*16+15
    int beg = eoff[n], end = beg + ecnt[n];
    float acc[16];
#pragma unroll
    for (int c = 0; c < 16; ++c) acc[c] = 0.f;
    float den = 0.f;
    for (int i = beg; i < end; ++i) {
      int sl = eL[i];
      float wgt = eE[2 * i + (q >> 1)];   // head = channel>>5 = q>>1
      den += wgt;
      const float* hrow = &R1[sl * 64 + q * 16];
#pragma unroll
      for (int c = 0; c < 16; ++c) acc[c] += hrow[c] * wgt;
    }
    float inv = 1.f / den;
#pragma unroll
    for (int c = 0; c < 16; ++c)
      R2[n * 64 + q * 16 + c] = acc[c] * inv + A.gatB[q * 16 + c];
  }
  __syncthreads();

  // 7. LN1(h1) -> f16 A fragments
  if (tid < 64) {
    float m = 0.f;
    for (int c = 0; c < 64; ++c) m += R2[tid * 64 + c];
    m *= (1.f / 64.f);
    float v = 0.f;
    for (int c = 0; c < 64; ++c) { float d = R2[tid * 64 + c] - m; v += d * d; }
    mu[tid] = m; rs[tid] = rsqrtf(v * (1.f / 64.f) + EPS_);
  }
  __syncthreads();
  for (int i = tid; i < 4096; i += 256) {
    int n = i >> 6, ch = i & 63;
    Ah[i] = (_Float16)((R2[i] - mu[n]) * rs[n] * A.ln1w[ch] + A.ln1b[ch]);
  }
  __syncthreads();

  // 8. FFN in two 64-wide halves: t = relu(hn@W1h+b1h) (R0); u += t@W2h (R1)
  for (int half = 0; half < 2; ++half) {
    for (int t = wave; t < 16; t += 8) {
      int tr = (t >> 2) << 4, tc = (t & 3) << 4;
      v8f c = {};
      for (int kk = 0; kk < 64; kk += 32) {
        v16h af = ld_a_pk(Ah, tr, kk, 64, lane);
        v16h bf = ld_b_pk(A.f1Wh, kk, half * 64 + tc, 64, lane);
        c = wmma_f16(af, bf, c);
      }
      st_c_lds(R0, c, tr, tc, 64, lane);
    }
    __syncthreads();
    for (int i = tid; i < 4096; i += 256) {
      float v = R0[i] + A.f1B[half * 64 + (i & 63)];
      R0[i] = v > 0.f ? v : 0.f;
    }
    __syncthreads();
    for (int t = wave; t < 16; t += 8) {
      int tr = (t >> 2) << 4, tc = (t & 3) << 4;
      v8f c = {};
      if (half) c = ld_c_lds(R1, tr, tc, 64, lane);
      for (int kk = 0; kk < 64; kk += 32) {
        v16h af = ld_a_lds_f32(R0, tr, kk, 64, lane);
        v16h bf = ld_b_pk(A.f2Wh, half * 64 + kk, tc, 128, lane);
        c = wmma_f16(af, bf, c);
      }
      st_c_lds(R1, c, tr, tc, 64, lane);
    }
    __syncthreads();
  }

  // 9. residual + bias, LN2, writeout (next-block input + cs columns)
  for (int i = tid; i < 4096; i += 256)
    R1[i] = R2[i] + R1[i] + A.f2B[i & 63];
  __syncthreads();
  if (tid < 64) {
    float m = 0.f;
    for (int c = 0; c < 64; ++c) m += R1[tid * 64 + c];
    m *= (1.f / 64.f);
    float v = 0.f;
    for (int c = 0; c < 64; ++c) { float d = R1[tid * 64 + c] - m; v += d * d; }
    mu[tid] = m; rs[tid] = rsqrtf(v * (1.f / 64.f) + EPS_);
  }
  __syncthreads();
  for (int i = tid; i < 4096; i += 256) {
    int n = i >> 6, ch = i & 63;
    float o = (R1[i] - mu[n]) * rs[n] * A.ln2w[ch] + A.ln2b[ch];
    A.x[(base + n) * 64 + ch] = o;
    A.cs[(base + n) * LAT_ + A.blk * 64 + ch] = o;
  }
}

// ---------------------------------------------------------------------------
// K2: GCN score heads, scale cs columns in place; emit f16 copy for lin3
// ---------------------------------------------------------------------------
struct ScoreArgs {
  float* cs; _Float16* csh;
  const int* esrc; const int* edst;
  const float* sW0; const float* sW1; const float* sW2;
  const float* sb0; const float* sb1; const float* sb2;
};

__global__ __launch_bounds__(128) void k_score(ScoreArgs A) {
  __shared__ float p[64], sacc[64], dinv[64];
  __shared__ int degi[64];
  const int g = blockIdx.x, tid = threadIdx.x, base = g * NPG;
  if (tid < 64) degi[tid] = 1;                 // self loop
  __syncthreads();
  for (int e = tid; e < EPG_; e += 128)
    atomicAdd(&degi[A.edst[g * EPG_ + e] - base], 1);
  __syncthreads();
  if (tid < 64) dinv[tid] = rsqrtf((float)degi[tid]);
  __syncthreads();
  const float* Ws[3] = {A.sW0, A.sW1, A.sW2};
  const float* bs[3] = {A.sb0, A.sb1, A.sb2};
  for (int j = 0; j < 3; ++j) {
    if (tid < 64) {
      float acc = 0.f;
      for (int ch = 0; ch < 64; ++ch) {
        float hv = A.cs[(base + tid) * LAT_ + j * 64 + ch];
        acc += (hv > 0.f ? hv : 0.f) * Ws[j][ch];
      }
      p[tid] = acc;
      sacc[tid] = acc * dinv[tid] * dinv[tid];  // self loop contribution
    }
    __syncthreads();
    for (int e = tid; e < EPG_; e += 128) {
      int sl = A.esrc[g * EPG_ + e] - base, dl = A.edst[g * EPG_ + e] - base;
      atomicAdd(&sacc[dl], p[sl] * dinv[sl] * dinv[dl]);
    }
    __syncthreads();
    const float bj = bs[j][0];
    for (int i = tid; i < 4096; i += 128) {
      int n = i >> 6, ch = i & 63;
      size_t idx = (size_t)(base + n) * LAT_ + j * 64 + ch;
      float v = A.cs[idx] * (sacc[n] + bj);
      A.cs[idx] = v;
      A.csh[idx] = (_Float16)v;
    }
    __syncthreads();
  }
}

// ---------------------------------------------------------------------------
// K3: sort-pool + conv1 + maxpool + conv2 + lin1, one workgroup per graph
// ---------------------------------------------------------------------------
struct PoolArgs {
  const float* cs; float* zb;
  const float* c1W; const float* c1B;
  const float* c2W; const float* c2B;
  const float* lW;  const float* lB;
};

__global__ __launch_bounds__(256) void k_pool(PoolArgs A) {
  __shared__ float Zs[KPOOL * LAT_];   // 5760
  __shared__ float o1[16 * 30];
  __shared__ float po[16 * 15];
  __shared__ float o2[32 * 11];
  __shared__ float key[64];
  __shared__ int sel[KPOOL];
  const int g = blockIdx.x, tid = threadIdx.x, base = g * NPG;
  if (tid < 64) key[tid] = A.cs[(base + tid) * LAT_ + (LAT_ - 1)];
  __syncthreads();
  if (tid < 64) {                       // stable descending rank
    float kv = key[tid];
    int r = 0;
    for (int j = 0; j < 64; ++j)
      r += (key[j] > kv) || (key[j] == kv && j < tid);
    if (r < KPOOL) sel[r] = tid;
  }
  __syncthreads();
  for (int i = tid; i < KPOOL * LAT_; i += 256) {
    int pos = i / LAT_, c = i - pos * LAT_;
    Zs[i] = A.cs[(base + sel[pos]) * LAT_ + c];
  }
  __syncthreads();
  for (int i = tid; i < 480; i += 256) {           // conv1, stride LAT
    int c = i / 30, kp = i - c * 30;
    float acc = A.c1B[c];
    for (int t = 0; t < LAT_; ++t) acc += A.c1W[c * LAT_ + t] * Zs[kp * LAT_ + t];
    o1[c * 30 + kp] = acc > 0.f ? acc : 0.f;
  }
  __syncthreads();
  if (tid < 240) {                                 // maxpool 2,2
    int c = tid / 15, x = tid - c * 15;
    float a0 = o1[c * 30 + 2 * x], a1 = o1[c * 30 + 2 * x + 1];
    po[tid] = a0 > a1 ? a0 : a1;
  }
  __syncthreads();
  for (int i = tid; i < 352; i += 256) {           // conv2
    int c = i / 11, x = i - c * 11;
    float acc = A.c2B[c];
    for (int ic = 0; ic < 16; ++ic)
      for (int t = 0; t < 5; ++t)
        acc += A.c2W[(c * 16 + ic) * 5 + t] * po[ic * 15 + x + t];
    o2[i] = acc > 0.f ? acc : 0.f;
  }
  __syncthreads();
  if (tid < 128) {                                 // lin1
    float acc = A.lB[tid];
    for (int i = 0; i < 352; ++i) acc += o2[i] * A.lW[i * 128 + tid];
    A.zb[g * 128 + tid] = acc > 0.f ? acc : 0.f;
  }
}

// ---------------------------------------------------------------------------
// K4: csl = cs @ lin3_W + b  (M=NT, K=192, N=128): all-f16 WMMA operands
// ---------------------------------------------------------------------------
__global__ __launch_bounds__(256) void k_lin3(const _Float16* __restrict__ csh,
                                              const _Float16* __restrict__ Wt,
                                              const float* __restrict__ b,
                                              float* __restrict__ csl) {
  const int g = blockIdx.x, tid = threadIdx.x;
  const int wave = tid >> 5, lane = tid & 31;
  const int row0 = g * NPG + ((wave >> 1) << 4);
  const int cb = (wave & 1) * 64;
  v8f c[4];
  for (int j = 0; j < 4; ++j)
    for (int q = 0; q < 8; ++q) c[j][q] = 0.f;
  for (int kk = 0; kk < LAT_; kk += 32) {
    v16h af = ld_a_pk(csh, row0, kk, LAT_, lane);
#pragma unroll
    for (int j = 0; j < 4; ++j) {
      v16h bf = ld_b_pk(Wt, kk, cb + 16 * j, LAT_, lane);
      c[j] = wmma_f16(af, bf, c[j]);
    }
  }
  const int cc0 = lane & 15, rb = (lane >> 4) << 3;
  for (int j = 0; j < 4; ++j) {
    int col = cb + 16 * j + cc0;
    for (int v = 0; v < 8; ++v)
      csl[(size_t)(row0 + rb + v) * 128 + col] = c[j][v] + b[col];
  }
}

// ---------------------------------------------------------------------------
// K5: cross-attention readout per graph
// ---------------------------------------------------------------------------
__global__ __launch_bounds__(256) void k_attn(const float* __restrict__ csl,
                                              const float* __restrict__ zb,
                                              const float* __restrict__ pW,
                                              const float* __restrict__ pB,
                                              float* __restrict__ f) {
  __shared__ float prj[128], sc[64], at[64];
  __shared__ float ssum;
  const int g = blockIdx.x, tid = threadIdx.x, base = g * NPG;
  if (tid < 128) {
    float acc = pB[tid];
    for (int i = 0; i < 128; ++i) acc += zb[g * 128 + i] * pW[i * 128 + tid];
    prj[tid] = acc;
  }
  __syncthreads();
  if (tid < 64) {
    float acc = 0.f;
    for (int c = 0; c < 128; ++c) acc += csl[(size_t)(base + tid) * 128 + c] * prj[c];
    sc[tid] = acc;
  }
  __syncthreads();
  if (tid == 0) {
    float m = sc[0];
    for (int n = 1; n < 64; ++n) m = sc[n] > m ? sc[n] : m;
    float s = 0.f;
    for (int n = 0; n < 64; ++n) { at[n] = __expf(sc[n] - m); s += at[n]; }
    ssum = s;
  }
  __syncthreads();
  if (tid < 128) {
    float acc = 0.f;
    for (int n = 0; n < 64; ++n) acc += at[n] * csl[(size_t)(base + n) * 128 + tid];
    f[g * 128 + tid] = acc / ssum;
  }
}

// ---------------------------------------------------------------------------
// K6: 0.9*f1 + 0.1*f2 -> lin2
// ---------------------------------------------------------------------------
__global__ __launch_bounds__(128) void k_final(const float* __restrict__ f1,
                                               const float* __restrict__ f2,
                                               const float* __restrict__ w,
                                               const float* __restrict__ b,
                                               float* __restrict__ out) {
  __shared__ float red[128];
  const int g = blockIdx.x, tid = threadIdx.x;
  float v = 0.9f * f1[g * 128 + tid] + 0.1f * f2[g * 128 + tid];
  red[tid] = v * w[tid];
  __syncthreads();
  for (int s = 64; s > 0; s >>= 1) {
    if (tid < s) red[tid] += red[tid + s];
    __syncthreads();
  }
  if (tid == 0) out[g] = red[0] + b[0];
}

// ---------------------------------------------------------------------------
// Host launcher
// ---------------------------------------------------------------------------
// Input order: 0 x1, 1 edge_index1, 2 x2, 3 edge_index2, 4 batch, then
// params flattened in jax pytree order (dict keys sorted, lists in order):
//  5..40: blocks[b] {att_dst, att_src, ffn1_W, ffn1_b, ffn2_W, ffn2_b,
//                    gat_W, gat_b, ln1_b, ln1_w, ln2_b, ln2_w}, b = 0..2
//  41 conv1_W  42 conv1_b  43 conv2_W  44 conv2_b
//  45 init_ln_b 46 init_ln_w
//  47 lin12_W 48 lin12_b 49 lin1_W 50 lin1_b 51 lin2_W 52 lin2_b
//  53 lin3_W 54 lin3_b 55 prj_W 56 prj_b
//  57..59 score_W[0..2]  60..62 score_b[0..2]
extern "C" void kernel_launch(void* const* d_in, const int* in_sizes, int n_in,
                              void* d_out, int out_size, void* d_ws, size_t ws_size,
                              hipStream_t stream) {
  (void)in_sizes; (void)n_in; (void)out_size; (void)ws_size;
  const float* x1 = (const float*)d_in[0];
  const int*   e1 = (const int*)d_in[1];
  const float* x2 = (const float*)d_in[2];
  const int*   e2 = (const int*)d_in[3];
  auto P = [&](int i) { return (const float*)d_in[i]; };

  float* buf_x = (float*)d_ws;                        // [NT,64]
  float* cs    = buf_x + (size_t)NT_ * 64;            // [NT,192] f32
  float* csl   = cs    + (size_t)NT_ * LAT_;          // [NT,128]
  float* zb    = csl   + (size_t)NT_ * 128;           // [B,128]
  float* fbr   = zb    + (size_t)BGR * 128;           // [2,B,128]
  _Float16* csh = (_Float16*)(fbr + (size_t)2 * BGR * 128);  // [NT,192] f16
  _Float16* wpk = csh + (size_t)NT_ * LAT_;           // packed f16 weights
  // per-block packed: gatW (64x64) | f1W (128x64) | f2W (64x128) = 20480 halfs
  _Float16* lin3h = wpk + 3 * 20480;                  // 128 x 192
  const int E = BGR * EPG_;

  // pack all WMMA weights once (zero-padded, column-major f16)
  for (int blk = 0; blk < 3; ++blk) {
    const int pb = 5 + blk * 12;
    _Float16* gw = wpk + blk * 20480;
    k_packw<<<16, 256, 0, stream>>>(P(pb + 6), gw,                64, 64,
                                    blk == 0 ? INF_ : 64);
    k_packw<<<32, 256, 0, stream>>>(P(pb + 2), gw + 4096,         64, 128, 64);
    k_packw<<<32, 256, 0, stream>>>(P(pb + 4), gw + 4096 + 8192, 128, 64, 128);
  }
  k_packw<<<96, 256, 0, stream>>>(P(53), lin3h, LAT_, 128, LAT_);

  for (int br = 0; br < 2; ++br) {
    const float* x    = br ? x2 : x1;
    const int*   eidx = br ? e2 : e1;
    const float* linW = br ? P(47) : P(49);
    const float* linB = br ? P(48) : P(50);

    k_initln<<<BGR, 128, 0, stream>>>(x, P(46), P(45), buf_x);

    for (int blk = 0; blk < 3; ++blk) {
      const int pb = 5 + blk * 12;
      _Float16* gw = wpk + blk * 20480;
      GatArgs ga;
      ga.x = buf_x; ga.cs = cs;
      ga.esrc = eidx; ga.edst = eidx + E;
      ga.gatWh = gw; ga.f1Wh = gw + 4096; ga.f2Wh = gw + 4096 + 8192;
      ga.attS = P(pb + 1); ga.attD = P(pb + 0); ga.gatB = P(pb + 7);
      ga.ln1w = P(pb + 9); ga.ln1b = P(pb + 8);
      ga.f1B = P(pb + 3);  ga.f2B = P(pb + 5);
      ga.ln2w = P(pb + 11); ga.ln2b = P(pb + 10);
      ga.blk = blk;
      k_gat<<<BGR, 256, 0, stream>>>(ga);
    }

    ScoreArgs sca = { cs, csh, eidx, eidx + E,
                      P(57), P(58), P(59), P(60), P(61), P(62) };
    k_score<<<BGR, 128, 0, stream>>>(sca);

    PoolArgs pa = { cs, zb, P(41), P(42), P(43), P(44), linW, linB };
    k_pool<<<BGR, 256, 0, stream>>>(pa);

    k_lin3<<<BGR, 256, 0, stream>>>(csh, lin3h, P(54), csl);

    k_attn<<<BGR, 256, 0, stream>>>(csl, zb, P(55), P(56),
                                    fbr + (size_t)br * BGR * 128);
  }

  k_final<<<BGR, 128, 0, stream>>>(fbr, fbr + (size_t)BGR * 128,
                                   P(51), P(52), (float*)d_out);
}
